// mLSTM_88682484728411
// MI455X (gfx1250) — matrix operand order repaired and checked by reading, not compiled, not run on hardware
//
#include <hip/hip_runtime.h>
#include <stdint.h>

#define DBATCH 8
#define SEQ    1024
#define DIM    512
#define MROWS  (DBATCH * SEQ)   // 8192
#define KDIM   512
#define NQKV   1536

typedef __attribute__((ext_vector_type(16))) __bf16 v16bf;
typedef __attribute__((ext_vector_type(8)))  __bf16 v8bf;
typedef __attribute__((ext_vector_type(8)))  float  v8f;

union ABFrag { v16bf v; v8bf h[2]; };

// Async global->LDS copy path (gfx1250). Guarded so the fallback compiles on any pass.
#if defined(__AMDGCN__) && __has_builtin(__builtin_amdgcn_global_load_async_to_lds_b128) && \
    __has_builtin(__builtin_amdgcn_s_wait_asynccnt)
#define HAVE_ASYNC_LDS 1
#else
#define HAVE_ASYNC_LDS 0
#endif

#if HAVE_ASYNC_LDS
typedef int v4i_vs __attribute__((vector_size(16)));
typedef __attribute__((address_space(1))) v4i_vs* g_v4i_ptr;
typedef __attribute__((address_space(3))) v4i_vs* l_v4i_ptr;
#endif

__device__ __forceinline__ void wait_async() {
#if HAVE_ASYNC_LDS
    __builtin_amdgcn_s_wait_asynccnt(0);
#endif
}

__device__ __forceinline__ unsigned short f2bf_bits(float f) {
    unsigned u = __builtin_bit_cast(unsigned, f);
    unsigned r = u + 0x7FFFu + ((u >> 16) & 1u);   // round-to-nearest-even
    return (unsigned short)(r >> 16);
}

// ---------------- converts ----------------
__global__ void k_cvt(const float* __restrict__ src, unsigned short* __restrict__ dst, int n) {
    int i = blockIdx.x * 256 + threadIdx.x;
    if (i < n) dst[i] = f2bf_bits(src[i]);
}

// dst[n*512 + k] = W[k*stride + n0 + n]  (bf16), n in [0,N)
__global__ void k_cvt_wT(const float* __restrict__ W, int stride, int n0, int N,
                         unsigned short* __restrict__ dst) {
    int i = blockIdx.x * 256 + threadIdx.x;
    if (i >= N * KDIM) return;
    int n = i >> 9, k = i & 511;
    dst[i] = f2bf_bits(W[(size_t)k * stride + n0 + n]);
}

__global__ void k_pack_bias(const float* __restrict__ b_in, const float* __restrict__ b_ig,
                            const float* __restrict__ b_fg, const float* __restrict__ b_og,
                            float* __restrict__ bq, float* __restrict__ bg) {
    int i = blockIdx.x * 256 + threadIdx.x;
    if (i < 1536) {
        bq[i] = b_in[i];
        bg[i] = (i < 512) ? b_ig[i] : (i < 1024 ? b_fg[i - 512] : b_og[i - 1024]);
    }
}

// ---------------- bf16 WMMA GEMM ----------------
// A:  [M, 512] bf16 row-major.  Bt: [N, 512] bf16 row-major (i.e. B transposed).
// Block tile 128x128, 8 waves (2x4), wave tile 64x32 = 4x2 WMMA 16x16 tiles, K-step 32.
#define LDSW 40

__device__ __forceinline__ void load_tile(const unsigned short* __restrict__ g,
                                          int rowBase, int kt,
                                          unsigned short (&lds)[128][LDSW], int tid) {
    int r  = tid >> 1;
    int cs = (tid & 1) * 16;
    const unsigned short* gp = g + (size_t)(rowBase + r) * KDIM + kt * 32 + cs;
    unsigned short* lp = &lds[r][cs];
#if HAVE_ASYNC_LDS
    // Copy 32B per lane straight into LDS via the async copy engine (ASYNCcnt).
    __builtin_amdgcn_global_load_async_to_lds_b128(
        (g_v4i_ptr)gp, (l_v4i_ptr)lp, 0, 0);
    __builtin_amdgcn_global_load_async_to_lds_b128(
        (g_v4i_ptr)(gp + 8), (l_v4i_ptr)(lp + 8), 0, 0);
#else
    const uint4* src = (const uint4*)gp;
    uint4 d0 = src[0];
    uint4 d1 = src[1];
    *(uint4*)lp       = d0;
    *(uint4*)(lp + 8) = d1;
#endif
}

template <int MODE>   // 0: qkv (tanh on cols<1024, also emit v as bf16)  1: sigmoid  2: plain
__global__ __launch_bounds__(256) void k_gemm(const unsigned short* __restrict__ A,
                                              const unsigned short* __restrict__ Bt,
                                              const float* __restrict__ bias,
                                              float* __restrict__ outF,
                                              unsigned short* __restrict__ outBF,
                                              int ldo) {
    __shared__ __align__(16) unsigned short As[2][128][LDSW];
    __shared__ __align__(16) unsigned short Bs[2][128][LDSW];

    const int tid   = threadIdx.x;
    const int lane  = tid & 31;
    const int wv    = tid >> 5;
    const int waveM = wv >> 2;      // 0..1
    const int waveN = wv & 3;       // 0..3
    const int l15   = lane & 15;
    const int kh    = lane >> 4;    // lane half

    const int mBase = blockIdx.y * 128;
    const int nBase = blockIdx.x * 128;

    const v8f zero = {0.f, 0.f, 0.f, 0.f, 0.f, 0.f, 0.f, 0.f};
    v8f acc[4][2];
#pragma unroll
    for (int mi = 0; mi < 4; ++mi)
#pragma unroll
        for (int ni = 0; ni < 2; ++ni) acc[mi][ni] = zero;

    load_tile(A,  mBase, 0, As[0], tid);
    load_tile(Bt, nBase, 0, Bs[0], tid);
    wait_async();
    __syncthreads();

    int buf = 0;
    const int KT = KDIM / 32;   // 16
    for (int kt = 0; kt < KT; ++kt) {
        if (kt + 1 < KT) {
            load_tile(A,  mBase, kt + 1, As[buf ^ 1], tid);
            load_tile(Bt, nBase, kt + 1, Bs[buf ^ 1], tid);
        }
        ABFrag a[4], b[2];
#pragma unroll
        for (int mi = 0; mi < 4; ++mi) {
            int r = waveM * 64 + mi * 16 + l15;
            // A layout: lanes 0-15 hold K 0..7 / 16..23; lanes 16-31 hold K 8..15 / 24..31
            a[mi].h[0] = *(const v8bf*)&As[buf][r][kh * 8];
            a[mi].h[1] = *(const v8bf*)&As[buf][r][16 + kh * 8];
        }
#pragma unroll
        for (int ni = 0; ni < 2; ++ni) {
            int c = waveN * 32 + ni * 16 + l15;
            // B layout: lanes 0-15 hold K 0..7 / 16..23; lanes 16-31 hold K 8..15 / 24..31
            b[ni].h[0] = *(const v8bf*)&Bs[buf][c][kh * 8];
            b[ni].h[1] = *(const v8bf*)&Bs[buf][c][16 + kh * 8];
        }
#pragma unroll
        for (int mi = 0; mi < 4; ++mi)
#pragma unroll
            for (int ni = 0; ni < 2; ++ni)
                acc[mi][ni] = __builtin_amdgcn_wmma_f32_16x16x32_bf16(
                    false, a[mi].v, false, b[ni].v, (short)0, acc[mi][ni], false, false);
        wait_async();        // next-buffer async copies (this wave's) landed in LDS
        __syncthreads();
        buf ^= 1;
    }

    // epilogue: C/D layout: lane n = lane&15, vgpr r -> m = r + 8*(lane>>4)
#pragma unroll
    for (int mi = 0; mi < 4; ++mi) {
#pragma unroll
        for (int ni = 0; ni < 2; ++ni) {
            const int col = nBase + waveN * 32 + ni * 16 + l15;
            const float bb = bias[col];
#pragma unroll
            for (int r = 0; r < 8; ++r) {
                int row = mBase + waveM * 64 + mi * 16 + r + (kh << 3);
                float v_ = acc[mi][ni][r] + bb;
                if (MODE == 0) {
                    if (col < 1024) v_ = tanhf(v_);
                    outF[(size_t)row * ldo + col] = v_;
                    if (col >= 1024) outBF[(size_t)row * 512 + (col - 1024)] = f2bf_bits(v_);
                } else if (MODE == 1) {
                    v_ = 1.0f / (1.0f + expf(-v_));
                    outF[(size_t)row * ldo + col] = v_;
                } else {
                    outF[(size_t)row * ldo + col] = v_;
                }
            }
        }
    }
}

// ---------------- recurrent scan ----------------
// grid (8 col-blocks, 8 batches), 256 threads. Each block owns C[512 x 64] in VGPRs
// (128 f32/thread). thread: e = tid&63 (column), stripe = tid>>6 (d range of 128).
__global__ __launch_bounds__(256) void k_scan(const float* __restrict__ qkv,   // [8192,1536] q|k|v
                                              const float* __restrict__ ifo,   // [8192,1536] i|f|o
                                              unsigned short* __restrict__ hb, // [8192,512] bf16 h
                                              float* __restrict__ outC,        // [8,512,512]
                                              float* __restrict__ outN,        // [8,512]
                                              float* __restrict__ outHlast) {  // [8,512]
    __shared__ float4 g4[512];          // {f, i*k, q, 0} per d
    __shared__ float  n_s[512];
    __shared__ float  red[256];
    __shared__ float  v_s[64], o_s[64];

    const int tid    = threadIdx.x;
    const int b      = blockIdx.y;
    const int e0     = blockIdx.x * 64;
    const int e      = tid & 63;
    const int stripe = tid >> 6;
    const int dbase  = stripe * 128;

    float C[128];
#pragma unroll
    for (int j = 0; j < 128; ++j) C[j] = 0.0f;
    for (int d = tid; d < 512; d += 256) n_s[d] = 0.0f;
    __syncthreads();

    for (int t = 0; t < SEQ; ++t) {
        const size_t base = ((size_t)b * SEQ + t) * 1536;
        float part = 0.0f;
        for (int d = tid; d < 512; d += 256) {
            float qd = qkv[base + d];
            float kd = qkv[base + 512 + d];
            float id = ifo[base + d];
            float fd = ifo[base + 512 + d];
            float nn = fd * n_s[d] + id * kd;
            n_s[d]   = nn;
            g4[d]    = make_float4(fd, id * kd, qd, 0.0f);
            part += qd * nn;
            // pull next timestep's gate rows toward the WGP while this step computes:
            // these loads sit directly on the 1024-step serial dependence chain.
            __builtin_prefetch(&qkv[base + 1536 + d], 0, 0);
            __builtin_prefetch(&ifo[base + 1536 + d], 0, 0);
        }
        if (tid < 64) {
            v_s[tid] = qkv[base + 1024 + e0 + tid];
            o_s[tid] = ifo[base + 1024 + e0 + tid];
        }
        red[tid] = part;
        __syncthreads();
        for (int s2 = 128; s2 > 0; s2 >>= 1) {
            if (tid < s2) red[tid] += red[tid + s2];
            __syncthreads();
        }
        const float denom = fmaxf(red[0], 1e-6f);

        const float vv = v_s[e];
        float h = 0.0f;
#pragma unroll
        for (int j = 0; j < 128; ++j) {
            float4 g = g4[dbase + j];
            float  c = fmaf(g.x, C[j], g.y * vv);   // C = f*C + (i*k)*v
            C[j] = c;
            h = fmaf(g.z, c, h);                    // h += q*C_new
        }
        __syncthreads();          // all threads done reading red[0]/g4 before reuse
        red[tid] = h;
        __syncthreads();
        if (tid < 64) {
            float ht = red[tid] + red[tid + 64] + red[tid + 128] + red[tid + 192];
            float hv = o_s[tid] * ht / denom;
            size_t ho = ((size_t)b * SEQ + t) * 512 + e0 + tid;
            hb[ho] = f2bf_bits(hv);
            if (t == SEQ - 1) outHlast[(size_t)b * 512 + e0 + tid] = hv;
        }
        __syncthreads();
    }

    // final state C -> d_out
#pragma unroll
    for (int j = 0; j < 128; ++j)
        outC[(((size_t)b * 512) + dbase + j) * 512 + e0 + e] = C[j];
    if (e0 == 0)
        for (int d = tid; d < 512; d += 256) outN[(size_t)b * 512 + d] = n_s[d];
}

// ---------------- host ----------------
extern "C" void kernel_launch(void* const* d_in, const int* in_sizes, int n_in,
                              void* d_out, int out_size, void* d_ws, size_t ws_size,
                              hipStream_t stream) {
    (void)in_sizes; (void)n_in; (void)out_size; (void)ws_size;
    const float* x     = (const float*)d_in[0];
    const float* W_in  = (const float*)d_in[1];
    const float* b_in  = (const float*)d_in[2];
    const float* W_ig  = (const float*)d_in[3];
    const float* b_ig  = (const float*)d_in[4];
    const float* W_fg  = (const float*)d_in[5];
    const float* b_fg  = (const float*)d_in[6];
    const float* W_og  = (const float*)d_in[7];
    const float* b_og  = (const float*)d_in[8];
    const float* W_out = (const float*)d_in[9];
    const float* b_out = (const float*)d_in[10];
    float* out = (float*)d_out;

    char* ws = (char*)d_ws;
    size_t off = 0;
    auto take = [&](size_t bytes) -> char* {
        char* p = ws + off;
        off = (off + bytes + 255) & ~(size_t)255;
        return p;
    };
    unsigned short* xb    = (unsigned short*)take((size_t)MROWS * KDIM * 2);
    unsigned short* WqkvT = (unsigned short*)take((size_t)NQKV * KDIM * 2);
    unsigned short* WgT   = (unsigned short*)take((size_t)NQKV * KDIM * 2);
    unsigned short* WoT   = (unsigned short*)take((size_t)DIM * KDIM * 2);
    float*          qkv   = (float*)take((size_t)MROWS * NQKV * 4);
    float*          ifo   = (float*)take((size_t)MROWS * NQKV * 4);
    unsigned short* vb    = (unsigned short*)take((size_t)MROWS * DIM * 2);
    unsigned short* hb    = (unsigned short*)take((size_t)MROWS * DIM * 2);
    float*          bq    = (float*)take(1536 * 4);
    float*          bg    = (float*)take(1536 * 4);

    float* outMain  = out;                                 // [8,1024,512]
    float* outC     = out + (size_t)DBATCH * SEQ * DIM;    // [8,512,512]
    float* outN     = outC + (size_t)DBATCH * DIM * DIM;   // [8,512]
    float* outHlast = outN + (size_t)DBATCH * DIM;         // [8,512]

    k_cvt<<<(MROWS * KDIM + 255) / 256, 256, 0, stream>>>(x, xb, MROWS * KDIM);
    k_cvt_wT<<<(NQKV * KDIM + 255) / 256, 256, 0, stream>>>(W_in, 2048, 0, NQKV, WqkvT);
    k_cvt_wT<<<(DIM * KDIM + 255) / 256, 256, 0, stream>>>(W_ig, 512, 0, DIM, WgT);
    k_cvt_wT<<<(DIM * KDIM + 255) / 256, 256, 0, stream>>>(W_fg, 512, 0, DIM, WgT + (size_t)512 * KDIM);
    k_cvt_wT<<<(DIM * KDIM + 255) / 256, 256, 0, stream>>>(W_og, 512, 0, DIM, WgT + (size_t)1024 * KDIM);
    k_cvt_wT<<<(DIM * KDIM + 255) / 256, 256, 0, stream>>>(W_out, 512, 0, DIM, WoT);
    k_pack_bias<<<6, 256, 0, stream>>>(b_in, b_ig, b_fg, b_og, bq, bg);

    dim3 blk(256);
    // proj -> q(tanh), k(tanh), v (+ v in bf16)
    k_gemm<0><<<dim3(NQKV / 128, MROWS / 128), blk, 0, stream>>>(xb, WqkvT, bq, qkv, vb, NQKV);
    // gates -> sigmoid(i|f|o)
    k_gemm<1><<<dim3(NQKV / 128, MROWS / 128), blk, 0, stream>>>(vb, WgT, bg, ifo, nullptr, NQKV);
    // sequential mLSTM scan, C resident in VGPRs
    k_scan<<<dim3(8, DBATCH), blk, 0, stream>>>(qkv, ifo, hb, outC, outN, outHlast);
    // out = hs @ W_out + b_out
    k_gemm<2><<<dim3(DIM / 128, MROWS / 128), blk, 0, stream>>>(hb, WoT, b_out, outMain, nullptr, DIM);
}
